// Item_GraphConvolution_mid_attention_57698590654950
// MI455X (gfx1250) — matrix operand order repaired and checked by reading, not compiled, not run on hardware
//
#include <hip/hip_runtime.h>
#include <hip/hip_bf16.h>

// ---------------------------------------------------------------------------
// Types for CDNA5 WMMA (wave32): A/B = 16 halves per lane, C/D = 8 floats.
// ---------------------------------------------------------------------------
typedef __attribute__((ext_vector_type(4)))  _Float16 v4h;
typedef __attribute__((ext_vector_type(8)))  _Float16 v8h;
typedef __attribute__((ext_vector_type(16))) _Float16 v16h;
typedef __attribute__((ext_vector_type(8)))  float    v8f;

__device__ __forceinline__ v16h frag16(const _Float16* p0, const _Float16* p1) {
    v8h lo = *(const v8h*)p0;
    v8h hi = *(const v8h*)p1;
    return __builtin_shufflevector(lo, hi, 0, 1, 2, 3, 4, 5, 6, 7,
                                           8, 9, 10, 11, 12, 13, 14, 15);
}

// Async global->LDS 16B copy (gfx1250 GLOBAL_LOAD_ASYNC_TO_LDS_B128, ASYNCcnt).
__device__ __forceinline__ void async_copy16(uint32_t lds_byte_off, uint64_t gaddr) {
    asm volatile("global_load_async_to_lds_b128 %0, %1, off"
                 :: "v"(lds_byte_off), "v"(gaddr) : "memory");
}
__device__ __forceinline__ void wait_async0() {
    asm volatile("s_wait_asynccnt 0x0" ::: "memory");
}
// Low 32 bits of a generic pointer to LDS == byte offset within LDS.
__device__ __forceinline__ uint32_t lds_off(const void* p) {
    return (uint32_t)(uintptr_t)p;
}

// ---------------------------------------------------------------------------
// WMMA GEMM:  C[M,128] = (A16[M,K] @ B16^T) (+ bscale*bias) (relu?)
//   A16: [M][K] f16 row-major; B16: [128][K] f16 row-major (n-major) so the
//   LDS B tile Bs[n][k] is a straight copy -> pure async b128 staging.
// Block tile 64x128, 256 thr = 8 waves; wave w owns cols [16w,16w+16),
// loops 4 M-subtiles; K consumed in 32-wide chunks (v_wmma_f32_16x16x32_f16).
// ---------------------------------------------------------------------------
#define GEMM_ROWS 64
#define APITCH 40   // halves -> 80B rows, chunks at 0/16/32/48B, 16B aligned
#define BPITCH 40

template <bool RELU, bool HASBIAS>
__global__ void __launch_bounds__(256)
gemm_f16_wmma(const _Float16* __restrict__ A16, const _Float16* __restrict__ B16,
              const float* __restrict__ bias, float bscale,
              float* __restrict__ C, int M, int K) {
    __shared__ __align__(16) _Float16 As[GEMM_ROWS * APITCH];
    __shared__ __align__(16) _Float16 Bs[128 * BPITCH];

    const int tid  = threadIdx.x;
    const int lane = tid & 31;
    const int wave = tid >> 5;
    const int row0 = blockIdx.x * GEMM_ROWS;

    // --- per-thread async-copy coordinates (16B granules) ---
    // As tile: 64 rows x 64B = 256 granules -> 1/thread. Clamp tail rows to
    // M-1 (their results are never stored) -> zero divergence.
    const int am = tid >> 2, ac = tid & 3;
    int arow = row0 + am; if (arow >= M) arow = M - 1;
    const uint64_t aga  = (uint64_t)(uintptr_t)(A16 + (size_t)arow * K + ac * 8);
    const uint32_t alds = lds_off(As) + (uint32_t)(am * APITCH + ac * 8) * 2;
    // Bs tile: 128 rows x 64B = 512 granules -> 2/thread.
    const int bn0 = tid >> 2;
    const int bn1 = (tid + 256) >> 2;
    const uint64_t bga0 = (uint64_t)(uintptr_t)(B16 + (size_t)bn0 * K + ac * 8);
    const uint64_t bga1 = (uint64_t)(uintptr_t)(B16 + (size_t)bn1 * K + ac * 8);
    const uint32_t blds0 = lds_off(Bs) + (uint32_t)(bn0 * BPITCH + ac * 8) * 2;
    const uint32_t blds1 = lds_off(Bs) + (uint32_t)(bn1 * BPITCH + ac * 8) * 2;

    v8f acc[4];
#pragma unroll
    for (int t = 0; t < 4; ++t)
#pragma unroll
        for (int i = 0; i < 8; ++i) acc[t][i] = 0.0f;

    const int kbA = (lane < 16) ? 0 : 8;   // ISA 7.12.2 A layout split
    const int kbB = (lane < 16) ? 0 : 8;   // ISA 7.12.2 B layout split

    for (int k0 = 0; k0 < K; k0 += 32) {
        __syncthreads();                       // LDS free to overwrite
        const uint64_t koff = (uint64_t)k0 * 2;
        async_copy16(alds,  aga  + koff);
        async_copy16(blds0, bga0 + koff);
        async_copy16(blds1, bga1 + koff);
        wait_async0();                         // own wave's copies done
        __syncthreads();                       // everyone's copies visible

        const _Float16* bp = &Bs[(16 * wave + (lane & 15)) * BPITCH];
        v16h bfrag = frag16(bp + kbB, bp + kbB + 16);
#pragma unroll
        for (int ms = 0; ms < 4; ++ms) {
            const _Float16* ap = &As[(16 * ms + (lane & 15)) * APITCH];
            v16h afrag = frag16(ap + kbA, ap + kbA + 16);
            acc[ms] = __builtin_amdgcn_wmma_f32_16x16x32_f16(
                false, afrag, false, bfrag, (short)0, acc[ms], false, false);
        }
    }

    // Epilogue: C/D layout -> VGPR i holds row (lane<16 ? i : 8+i).
    // Lanes 0-15 / 16-31 each store one contiguous 64B row segment per op.
    const int ncol = 16 * wave + (lane & 15);
    const int rbase = (lane < 16) ? 0 : 8;
    float badd = HASBIAS ? bscale * bias[ncol] : 0.0f;

    if (row0 + GEMM_ROWS <= M) {
        // Fast path (all blocks except the single tail block): no guards;
        // one base address per tile, rows step by a constant 512B.
        float* cp = C + (size_t)(row0 + rbase) * 128 + ncol;
#pragma unroll
        for (int ms = 0; ms < 4; ++ms) {
            float* cpt = cp + (size_t)ms * 16 * 128;
#pragma unroll
            for (int i = 0; i < 8; ++i) {
                float v = acc[ms][i] + badd;
                if (RELU) v = fmaxf(v, 0.0f);
                cpt[(size_t)i * 128] = v;
            }
        }
    } else {
#pragma unroll
        for (int ms = 0; ms < 4; ++ms) {
#pragma unroll
            for (int i = 0; i < 8; ++i) {
                int m = row0 + 16 * ms + rbase + i;
                if (m < M) {
                    float v = acc[ms][i] + badd;
                    if (RELU) v = fmaxf(v, 0.0f);
                    C[(size_t)m * 128 + ncol] = v;
                }
            }
        }
    }
}

// ---------------------------------------------------------------------------
// SpMM: one wave32 per edge; lane carries float4 (32*4 = 128 = E).
// support/Ax are 51.2 MB each -> resident in 192 MB L2: L2 gather + f32 atomics.
// ---------------------------------------------------------------------------
__global__ void __launch_bounds__(256)
spmm_kernel(const float* __restrict__ X, const int* __restrict__ erow,
            const int* __restrict__ ecol, const float* __restrict__ evals,
            float* __restrict__ Y, int nE) {
    int e = (int)(blockIdx.x * 8u + (threadIdx.x >> 5));
    if (e >= nE) return;
    int lane = threadIdx.x & 31;
    int r = erow[e], c = ecol[e];
    float v = evals[e];
    float4 x = ((const float4*)(X + (size_t)c * 128))[lane];
    float* yp = Y + (size_t)r * 128 + lane * 4;
    unsafeAtomicAdd(yp + 0, x.x * v);   // -> global_atomic_add_f32
    unsafeAtomicAdd(yp + 1, x.y * v);
    unsafeAtomicAdd(yp + 2, x.z * v);
    unsafeAtomicAdd(yp + 3, x.w * v);
}

// ---------------------------------------------------------------------------
// Elementwise prep kernels (one-shot; ~130MB extra HBM traffic ~= 6us at
// 23.3 TB/s, trivial vs the ~3.3GB L2-resident SpMM traffic).
// ---------------------------------------------------------------------------
__global__ void fill_zero_kernel(float4* __restrict__ p, long long n4) {
    long long i = (long long)blockIdx.x * blockDim.x + threadIdx.x;
    if (i < n4) p[i] = make_float4(0.f, 0.f, 0.f, 0.f);
}

__global__ void cvt_f16_kernel(const float* __restrict__ src,
                               _Float16* __restrict__ dst, long long n) {
    long long i = ((long long)blockIdx.x * blockDim.x + threadIdx.x) * 4;
    if (i < n) {  // n is a multiple of 4 (rows of 128/256)
        float4 v = *(const float4*)(src + i);
        v4h h = { (_Float16)v.x, (_Float16)v.y, (_Float16)v.z, (_Float16)v.w };
        *(v4h*)(dst + i) = h;
    }
}

// WT16[n*K + k] = (f16) W[k*128 + n]   (weight is [K,128] k-major)
__global__ void wtrans_f16_kernel(const float* __restrict__ W,
                                  _Float16* __restrict__ WT, int K) {
    int i = blockIdx.x * 256 + threadIdx.x;
    if (i < 128 * K) {
        int n = i / K, k = i % K;
        WT[i] = (_Float16)W[(size_t)k * 128 + n];
    }
}

// Y16 = (f16)(a + b)  -- fuses output_low+output_mid = Ax + A^2x
__global__ void addcvt_f16_kernel(const float* __restrict__ a,
                                  const float* __restrict__ b,
                                  _Float16* __restrict__ dst, long long n) {
    long long i = ((long long)blockIdx.x * blockDim.x + threadIdx.x) * 4;
    if (i < n) {
        float4 va = *(const float4*)(a + i);
        float4 vb = *(const float4*)(b + i);
        v4h h = { (_Float16)(va.x + vb.x), (_Float16)(va.y + vb.y),
                  (_Float16)(va.z + vb.z), (_Float16)(va.w + vb.w) };
        *(v4h*)(dst + i) = h;
    }
}

// ---------------------------------------------------------------------------
// out = (Ax + A2x) @ Wv^T + 2*bv
// (softmax over the q-axis makes each column sum to 1, and context.sum(q)
//  then collapses the attention to exactly v0+v1 -> Q/K branches are dead.)
// ---------------------------------------------------------------------------
extern "C" void kernel_launch(void* const* d_in, const int* in_sizes, int n_in,
                              void* d_out, int out_size, void* d_ws, size_t ws_size,
                              hipStream_t stream) {
    (void)n_in; (void)out_size; (void)ws_size;
    const float* feature = (const float*)d_in[0];
    const int*   erow    = (const int*)d_in[1];
    const int*   ecol    = (const int*)d_in[2];
    const float* evals   = (const float*)d_in[3];
    const float* weight  = (const float*)d_in[4];   // [256,128] k-major
    const float* Wv      = (const float*)d_in[9];   // [128,128] torch [out,in]
    const float* bv      = (const float*)d_in[10];  // [128]

    const int F = 256, E = 128;
    const int N  = in_sizes[0] / F;
    const int NE = in_sizes[1];

    float*    support = (float*)d_ws;                    // [N,128] f32
    float*    Ax      = support + (size_t)N * E;         // [N,128] f32
    float*    Ax2     = Ax + (size_t)N * E;              // [N,128] f32
    _Float16* feat16  = (_Float16*)(Ax2 + (size_t)N * E);// [N,256] f16
    _Float16* Y16     = feat16 + (size_t)N * F;          // [N,128] f16
    _Float16* WT16    = Y16 + (size_t)N * E;             // [128,256] f16
    _Float16* Wv16    = WT16 + (size_t)128 * F;          // [128,128] f16

    // Prep: zero accumulators, convert/transpose operands to f16.
    long long nz4 = (long long)2 * N * E / 4;
    fill_zero_kernel<<<(unsigned)((nz4 + 255) / 256), 256, 0, stream>>>((float4*)Ax, nz4);
    long long nf = (long long)N * F;
    cvt_f16_kernel<<<(unsigned)((nf / 4 + 255) / 256), 256, 0, stream>>>(feature, feat16, nf);
    wtrans_f16_kernel<<<(128 * F + 255) / 256, 256, 0, stream>>>(weight, WT16, F);
    cvt_f16_kernel<<<(128 * E / 4 + 255) / 256, 256, 0, stream>>>(Wv, Wv16, 128 * E);

    dim3 gblk((N + GEMM_ROWS - 1) / GEMM_ROWS);
    // support = relu(feature @ weight)
    gemm_f16_wmma<true, false><<<gblk, 256, 0, stream>>>(
        feat16, WT16, nullptr, 0.0f, support, N, F);

    unsigned sblk = (unsigned)((NE + 7) / 8);
    spmm_kernel<<<sblk, 256, 0, stream>>>(support, erow, ecol, evals, Ax, NE);
    spmm_kernel<<<sblk, 256, 0, stream>>>(Ax, erow, ecol, evals, Ax2, NE);

    // Y = Ax + A^2x  (f16), then out = Y @ Wv^T + 2*bv
    long long ny = (long long)N * E;
    addcvt_f16_kernel<<<(unsigned)((ny / 4 + 255) / 256), 256, 0, stream>>>(Ax, Ax2, Y16, ny);
    gemm_f16_wmma<false, true><<<gblk, 256, 0, stream>>>(
        Y16, Wv16, bv, 2.0f, (float*)d_out, N, E);
}
